// RecurrentEncoderDecoder_3444563771482
// MI455X (gfx1250) — compile-verified, matrix operand order checked by reading, not run on hardware
//
#include <hip/hip_runtime.h>
#include <hip/hip_bf16.h>
#include <cstdint>

// ---------------------------------------------------------------------------
// Problem constants (from reference)
// ---------------------------------------------------------------------------
constexpr int Bc   = 128;           // batch
constexpr int Tc   = 512;           // encoder time steps
constexpr int HORc = 64;            // decoder horizon
constexpr int Mc   = 1;
constexpr int Qc   = 5;
constexpr int MQc  = Mc * Qc;       // 5
constexpr int DFFc = 7;
constexpr int DINc = DFFc + Mc;     // 8
constexpr int Hc   = 512;           // hidden
constexpr int Gc   = 4 * Hc;        // 2048 gates

typedef _Float16 v8h  __attribute__((ext_vector_type(8)));
typedef _Float16 v16h __attribute__((ext_vector_type(16)));
typedef float    v8f  __attribute__((ext_vector_type(8)));

// ---------------------------------------------------------------------------
// WMMA helpers (gfx1250, wave32). Fragment layouts per CDNA5 ISA 7.12.2:
//  A (16x32 f16, MxK): lane&15 = row M, lane>>4 selects K-half;
//    v16h elems 0..7  = K (sel*8 .. sel*8+7)
//    v16h elems 8..15 = K (16 + sel*8 .. 16 + sel*8 + 7)
//  B is loaded identically from the N x K row-major weight array (B = W^T),
//  with lane&15 = output column N.
//  C/D (16x16 f32): elem r -> row r + 8*(lane>>4), col = lane&15.
// ---------------------------------------------------------------------------
__device__ __forceinline__ v16h load_frag_f16(const _Float16* __restrict__ base,
                                              int rowBase, int rowStride, int k) {
  const int lane = threadIdx.x & 31;
  const int r    = lane & 15;
  const int sel  = (lane >> 4) & 1;
  const _Float16* p = base + (size_t)(rowBase + r) * (size_t)rowStride + k + sel * 8;
  v8h lo = *(const v8h*)(p);
  v8h hi = *(const v8h*)(p + 16);
  v16h out;
#pragma unroll
  for (int i = 0; i < 8; ++i) { out[i] = lo[i]; out[i + 8] = hi[i]; }
  return out;
}

__device__ __forceinline__ v8f wmma16x16x32(v16h a, v16h b, v8f c) {
  return __builtin_amdgcn_wmma_f32_16x16x32_f16(false, a, false, b,
                                                (short)0, c, false, false);
}

__device__ __forceinline__ float sigmoidf_(float x) {
  return 1.0f / (1.0f + __expf(-x));
}

// ---------------------------------------------------------------------------
// Fused LSTM step:  gates = hIn @ wHH^T  [+ xA @ wIH^T]  [+ xs @ wS^T] + bias
// then cell update, all in one kernel. Block = 256 thr (8 waves). Each block
// owns 16 hidden columns (gate cols n, n+H, n+2H, n+3H) for all 128 rows.
// gridDim.x = H/16, gridDim.y = #directions (dir offsets added per pointer).
// ---------------------------------------------------------------------------
struct StepArgs {
  const _Float16* hIn;   long long hInDirOff;
  const _Float16* wHH;   long long wHHDirOff;   // G x H  (row-major, f16)
  const _Float16* xA;    long long xADirOff;    // B x K2 (optional 2nd chain)
  int xARowStride;       int K2;
  const _Float16* wIH;   long long wIHDirOff;   // G x K2 (row-major, f16)
  const float*    xs;    long long xsDirOff;    // B x Ks (f32, scalar-FMA path)
  int xsRowStride;       int Ks;
  const float*    wS;    long long wSDirOff;    // G x Ks (f32)
  const float*    bias;  long long biasDirOff;  // G
  float*          cBuf;  long long cDirOff;     // B x H (f32, in-place)
  _Float16*       hOut;  long long hOutDirOff;  // B x H
  _Float16*       ysOut; long long ysDirOff;    // optional seq output (f16)
  int ysRowStride;
  float*          hOutF32; long long hOutF32DirOff; // optional f32 h output
};

__global__ __launch_bounds__(256) void lstm_step_kernel(StepArgs A) {
  const long long dir = blockIdx.y;
  const _Float16* hIn  = A.hIn + dir * A.hInDirOff;
  const _Float16* wHH  = A.wHH + dir * A.wHHDirOff;
  const float*    bias = A.bias + dir * A.biasDirOff;
  float*          cBuf = A.cBuf + dir * A.cDirOff;
  _Float16*       hOut = A.hOut + dir * A.hOutDirOff;
  const _Float16* xA   = A.xA  ? A.xA  + dir * A.xADirOff  : nullptr;
  const _Float16* wIH  = A.wIH ? A.wIH + dir * A.wIHDirOff : nullptr;
  const float*    xs   = A.xs  ? A.xs  + dir * A.xsDirOff  : nullptr;
  const float*    wS   = A.wS  ? A.wS  + dir * A.wSDirOff  : nullptr;
  _Float16*       ysO  = A.ysOut   ? A.ysOut   + dir * A.ysDirOff      : nullptr;
  float*          hO32 = A.hOutF32 ? A.hOutF32 + dir * A.hOutF32DirOff : nullptr;

  const int n0    = blockIdx.x * 16;         // hidden-column slice
  const int mBase = (threadIdx.x >> 5) * 16; // wave -> batch-row tile

  v8f acc[4] = {v8f{}, v8f{}, v8f{}, v8f{}}; // i, f, g, o gate tiles

  // Recurrent chain: h (B x H) x wHH^T
  for (int k = 0; k < Hc; k += 32) {
    v16h a = load_frag_f16(hIn, mBase, Hc, k);
#pragma unroll
    for (int g = 0; g < 4; ++g) {
      v16h b = load_frag_f16(wHH, g * Hc + n0, Hc, k);
      acc[g] = wmma16x16x32(a, b, acc[g]);
    }
  }
  // Optional input-projection chain: xA (B x K2) x wIH^T
  if (xA) {
    for (int k = 0; k < A.K2; k += 32) {
      v16h a = load_frag_f16(xA, mBase, A.xARowStride, k);
#pragma unroll
      for (int g = 0; g < 4; ++g) {
        v16h b = load_frag_f16(wIH, g * Hc + n0, A.K2, k);
        acc[g] = wmma16x16x32(a, b, acc[g]);
      }
    }
  }

  const int lane = threadIdx.x & 31;
  const int col  = lane & 15;
  const int rofs = (lane >> 4) * 8;
  const int n    = n0 + col;

#pragma unroll
  for (int r = 0; r < 8; ++r) {
    const int row = mBase + r + rofs;  // batch index
    float gi = acc[0][r] + bias[n];
    float gf = acc[1][r] + bias[Hc + n];
    float gg = acc[2][r] + bias[2 * Hc + n];
    float go = acc[3][r] + bias[3 * Hc + n];
    if (xs) {  // tiny-K (<=8) input features: scalar f32 FMAs
      for (int kk = 0; kk < A.Ks; ++kk) {
        const float xv = xs[(size_t)row * A.xsRowStride + kk];
        gi += xv * wS[(size_t)(n) * A.Ks + kk];
        gf += xv * wS[(size_t)(Hc + n) * A.Ks + kk];
        gg += xv * wS[(size_t)(2 * Hc + n) * A.Ks + kk];
        go += xv * wS[(size_t)(3 * Hc + n) * A.Ks + kk];
      }
    }
    const float iv = sigmoidf_(gi);
    const float fv = sigmoidf_(gf);
    const float gv = tanhf(gg);
    const float ov = sigmoidf_(go);
    const size_t idx = (size_t)row * Hc + n;
    const float cn = fv * cBuf[idx] + iv * gv;
    cBuf[idx] = cn;
    const float hv = ov * tanhf(cn);
    hOut[idx] = (_Float16)hv;
    if (ysO)  ysO[(size_t)row * A.ysRowStride + n] = (_Float16)hv;
    if (hO32) hO32[idx] = hv;
  }
}

// ---------------------------------------------------------------------------
// Generic 128 x N WMMA GEMM: out = act(A(128xK,f16) @ Bw(NxK,f16)^T + bias)
// act: 0 = none, 1 = ELU. Block = 256 thr (8 waves = 8 M-tiles), grid = N/16.
// ---------------------------------------------------------------------------
__global__ __launch_bounds__(256) void gemm_wmma_kernel(
    const _Float16* __restrict__ Am, const _Float16* __restrict__ Bw,
    const float* __restrict__ bias, float* __restrict__ out,
    int K, int N, int act) {
  const int n0    = blockIdx.x * 16;
  const int mBase = (threadIdx.x >> 5) * 16;
  v8f acc = v8f{};
  for (int k = 0; k < K; k += 32) {
    v16h a = load_frag_f16(Am, mBase, K, k);
    v16h b = load_frag_f16(Bw, n0, K, k);
    acc = wmma16x16x32(a, b, acc);
  }
  const int lane = threadIdx.x & 31;
  const int col  = lane & 15;
  const int rofs = (lane >> 4) * 8;
  const int n    = n0 + col;
#pragma unroll
  for (int r = 0; r < 8; ++r) {
    const int row = mBase + r + rofs;
    float v = acc[r] + (bias ? bias[n] : 0.0f);
    if (act == 1) v = (v > 0.0f) ? v : (__expf(v) - 1.0f);
    out[(size_t)row * N + n] = v;
  }
}

// ---------------------------------------------------------------------------
// Small utility kernels
// ---------------------------------------------------------------------------
__global__ void f32_to_f16_kernel(const float* __restrict__ s,
                                  _Float16* __restrict__ d, long long n) {
  long long i = (long long)blockIdx.x * blockDim.x + threadIdx.x;
  const long long stride = (long long)gridDim.x * blockDim.x;
  for (; i < n; i += stride) d[i] = (_Float16)s[i];
}

__global__ void pack_h16_kernel(const _Float16* s0, const _Float16* s1,
                                const _Float16* s2, const _Float16* s3,
                                _Float16* dst) {  // dst: B x 4H
  int i = blockIdx.x * blockDim.x + threadIdx.x;
  if (i >= Bc * Hc) return;
  int b = i / Hc, n = i % Hc;
  size_t o = (size_t)b * (4 * Hc) + n;
  dst[o] = s0[i]; dst[o + Hc] = s1[i];
  dst[o + 2 * Hc] = s2[i]; dst[o + 3 * Hc] = s3[i];
}

__global__ void pack_c16_kernel(const float* s0, const float* s1,
                                const float* s2, const float* s3,
                                _Float16* dst) {  // dst: B x 4H
  int i = blockIdx.x * blockDim.x + threadIdx.x;
  if (i >= Bc * Hc) return;
  int b = i / Hc, n = i % Hc;
  size_t o = (size_t)b * (4 * Hc) + n;
  dst[o] = (_Float16)s0[i]; dst[o + Hc] = (_Float16)s1[i];
  dst[o + 2 * Hc] = (_Float16)s2[i]; dst[o + 3 * Hc] = (_Float16)s3[i];
}

__global__ void dec_init_kernel(const float* __restrict__ X,
                                const float* __restrict__ h_m,
                                const float* __restrict__ c_m,
                                float* inp_y, _Float16* h0, float* c0,
                                _Float16* h1, float* c1) {
  int i = blockIdx.x * blockDim.x + threadIdx.x;
  if (i < Bc) inp_y[i] = X[(size_t)i * Tc * DINc + (size_t)(Tc - 1) * DINc + 0];
  if (i < Bc * Hc) {
    int b = i / Hc, n = i % Hc;
    h0[i] = (_Float16)h_m[(size_t)b * 1024 + n];
    c0[i] = c_m[(size_t)b * 1024 + n];
    h1[i] = (_Float16)h_m[(size_t)b * 1024 + Hc + n];
    c1[i] = c_m[(size_t)b * 1024 + Hc + n];
  }
}

__global__ void xdec_kernel(const float* __restrict__ inp_y,
                            const float* __restrict__ ff, int t,
                            float* __restrict__ xdec) {
  int i = blockIdx.x * blockDim.x + threadIdx.x;
  if (i >= Bc * DINc) return;
  int b = i >> 3, k = i & 7;
  xdec[i] = (k == 0) ? inp_y[b]
                     : ff[(size_t)b * HORc * DFFc + (size_t)t * DFFc + (k - 1)];
}

__global__ void proj_kernel(const float* __restrict__ h1,
                            const float* __restrict__ w_out,
                            const float* __restrict__ b_out,
                            float* __restrict__ out, float* __restrict__ inp_y,
                            int t) {
  int i = blockIdx.x * blockDim.x + threadIdx.x;  // B*MQ = 640
  if (i >= Bc * MQc) return;
  int b = i / MQc, q = i % MQc;
  float s = b_out[q];
  for (int k = 0; k < Hc; ++k) s += h1[(size_t)b * Hc + k] * w_out[(size_t)q * Hc + k];
  out[(size_t)b * HORc * MQc + (size_t)t * MQc + q] = s;
  if (q == 0) inp_y[b] = s;
}

// ---------------------------------------------------------------------------
// Host orchestration
// ---------------------------------------------------------------------------
extern "C" void kernel_launch(void* const* d_in, const int* in_sizes, int n_in,
                              void* d_out, int out_size, void* d_ws, size_t ws_size,
                              hipStream_t stream) {
  (void)in_sizes; (void)n_in; (void)out_size; (void)ws_size;
  const float* X           = (const float*)d_in[0];
  const float* FF          = (const float*)d_in[1];
  const float* enc_w_ih_l0 = (const float*)d_in[4];
  const float* enc_w_hh_l0 = (const float*)d_in[5];
  const float* enc_b_l0    = (const float*)d_in[6];
  const float* enc_w_ih_l1 = (const float*)d_in[7];
  const float* enc_w_hh_l1 = (const float*)d_in[8];
  const float* enc_b_l1    = (const float*)d_in[9];
  const float* dec_w_ih_l0 = (const float*)d_in[10];
  const float* dec_w_hh_l0 = (const float*)d_in[11];
  const float* dec_b_l0    = (const float*)d_in[12];
  const float* dec_w_ih_l1 = (const float*)d_in[13];
  const float* dec_w_hh_l1 = (const float*)d_in[14];
  const float* dec_b_l1    = (const float*)d_in[15];
  const float* w_bidi      = (const float*)d_in[16];
  const float* b_bidi      = (const float*)d_in[17];
  const float* w_out       = (const float*)d_in[18];
  const float* b_out       = (const float*)d_in[19];
  float* out = (float*)d_out;

  // --- workspace bump allocator ---
  char* base = (char*)d_ws; size_t off = 0;
  auto alloc = [&](size_t bytes) -> void* {
    void* r = base + off; off = (off + bytes + 255) & ~(size_t)255; return r;
  };
  const size_t BH = (size_t)Bc * Hc;
  _Float16* ys1_16     = (_Float16*)alloc((size_t)Tc * Bc * 2 * Hc * 2); // 134 MB
  _Float16* wE0hh16    = (_Float16*)alloc((size_t)2 * Gc * Hc * 2);
  _Float16* wE1hh16    = (_Float16*)alloc((size_t)2 * Gc * Hc * 2);
  _Float16* wE1ih16    = (_Float16*)alloc((size_t)2 * Gc * 2 * Hc * 2);
  _Float16* wD0hh16    = (_Float16*)alloc((size_t)Gc * Hc * 2);
  _Float16* wD1ih16    = (_Float16*)alloc((size_t)Gc * Hc * 2);
  _Float16* wD1hh16    = (_Float16*)alloc((size_t)Gc * Hc * 2);
  _Float16* wBidi16    = (_Float16*)alloc((size_t)1024 * 2048 * 2);
  _Float16* hEnc0      = (_Float16*)alloc(2 * 2 * BH * 2); // [dir][ping][B*H]
  _Float16* hEnc1      = (_Float16*)alloc(2 * 2 * BH * 2);
  float*    cEnc0      = (float*)alloc(2 * BH * 4);
  float*    cEnc1      = (float*)alloc(2 * BH * 4);
  _Float16* hFlat16    = (_Float16*)alloc((size_t)Bc * 4 * Hc * 2);
  _Float16* cFlat16    = (_Float16*)alloc((size_t)Bc * 4 * Hc * 2);
  float*    h_m        = (float*)alloc((size_t)Bc * 1024 * 4);
  float*    c_m        = (float*)alloc((size_t)Bc * 1024 * 4);
  _Float16* hDec0      = (_Float16*)alloc(2 * BH * 2);     // [ping][B*H]
  _Float16* hDec1      = (_Float16*)alloc(2 * BH * 2);
  float*    cDec0      = (float*)alloc(BH * 4);
  float*    cDec1      = (float*)alloc(BH * 4);
  float*    hDec1F32   = (float*)alloc(BH * 4);
  float*    xdec       = (float*)alloc((size_t)Bc * DINc * 4);
  float*    inp_y      = (float*)alloc((size_t)Bc * 4);

  // --- weight conversion to f16 ---
  auto cvt = [&](const float* s, _Float16* d, long long n) {
    int blocks = (int)((n + 255) / 256); if (blocks > 4096) blocks = 4096;
    f32_to_f16_kernel<<<blocks, 256, 0, stream>>>(s, d, n);
  };
  cvt(enc_w_hh_l0, wE0hh16, 2LL * Gc * Hc);
  cvt(enc_w_hh_l1, wE1hh16, 2LL * Gc * Hc);
  cvt(enc_w_ih_l1, wE1ih16, 2LL * Gc * 2 * Hc);
  cvt(dec_w_hh_l0, wD0hh16, (long long)Gc * Hc);
  cvt(dec_w_ih_l1, wD1ih16, (long long)Gc * Hc);
  cvt(dec_w_hh_l1, wD1hh16, (long long)Gc * Hc);
  cvt(w_bidi,      wBidi16, 1024LL * 2048);

  // --- zero initial states ---
  hipMemsetAsync(hEnc0, 0, 2 * 2 * BH * 2, stream);
  hipMemsetAsync(hEnc1, 0, 2 * 2 * BH * 2, stream);
  hipMemsetAsync(cEnc0, 0, 2 * BH * 4, stream);
  hipMemsetAsync(cEnc1, 0, 2 * BH * 4, stream);

  const dim3 stepGrid2(Hc / 16, 2), stepGrid1(Hc / 16, 1);

  // --- encoder layer 0 (both directions concurrently; bwd = time-mirrored) ---
  for (int t = 0; t < Tc; ++t) {
    const int cur = t & 1, nxt = 1 - cur;
    StepArgs a{};
    a.hIn  = hEnc0 + (size_t)cur * BH;  a.hInDirOff  = 2 * (long long)BH;
    a.wHH  = wE0hh16;                   a.wHHDirOff  = (long long)Gc * Hc;
    a.xA   = nullptr;                   a.wIH = nullptr; a.K2 = 0;
    a.xs   = X + (size_t)t * DINc;      a.xsDirOff   = (long long)(Tc - 1 - 2 * t) * DINc;
    a.xsRowStride = Tc * DINc;          a.Ks = DINc;
    a.wS   = enc_w_ih_l0;               a.wSDirOff   = (long long)Gc * DINc;
    a.bias = enc_b_l0;                  a.biasDirOff = Gc;
    a.cBuf = cEnc0;                     a.cDirOff    = (long long)BH;
    a.hOut = hEnc0 + (size_t)nxt * BH;  a.hOutDirOff = 2 * (long long)BH;
    a.ysOut = ys1_16 + (size_t)t * Bc * 2 * Hc;
    a.ysDirOff = (long long)(Tc - 1 - 2 * t) * Bc * 2 * Hc + Hc;
    a.ysRowStride = 2 * Hc;
    a.hOutF32 = nullptr;
    lstm_step_kernel<<<stepGrid2, 256, 0, stream>>>(a);
  }
  // --- encoder layer 1 (input projection fused as second WMMA chain) ---
  for (int t = 0; t < Tc; ++t) {
    const int cur = t & 1, nxt = 1 - cur;
    StepArgs a{};
    a.hIn  = hEnc1 + (size_t)cur * BH;  a.hInDirOff  = 2 * (long long)BH;
    a.wHH  = wE1hh16;                   a.wHHDirOff  = (long long)Gc * Hc;
    a.xA   = ys1_16 + (size_t)t * Bc * 2 * Hc;
    a.xADirOff = (long long)(Tc - 1 - 2 * t) * Bc * 2 * Hc;
    a.xARowStride = 2 * Hc;             a.K2 = 2 * Hc;
    a.wIH  = wE1ih16;                   a.wIHDirOff  = (long long)Gc * 2 * Hc;
    a.xs   = nullptr; a.wS = nullptr;   a.Ks = 0;
    a.bias = enc_b_l1;                  a.biasDirOff = Gc;
    a.cBuf = cEnc1;                     a.cDirOff    = (long long)BH;
    a.hOut = hEnc1 + (size_t)nxt * BH;  a.hOutDirOff = 2 * (long long)BH;
    a.ysOut = nullptr; a.hOutF32 = nullptr;
    lstm_step_kernel<<<stepGrid2, 256, 0, stream>>>(a);
  }

  // --- merge final states: h_flat = [hf0, hb0, hf1, hb1], c likewise ---
  // T even -> final h is in ping buffer 0.
  const int pk = (int)((BH + 255) / 256);
  pack_h16_kernel<<<pk, 256, 0, stream>>>(hEnc0, hEnc0 + 2 * BH,
                                          hEnc1, hEnc1 + 2 * BH, hFlat16);
  pack_c16_kernel<<<pk, 256, 0, stream>>>(cEnc0, cEnc0 + BH,
                                          cEnc1, cEnc1 + BH, cFlat16);
  // h_m / c_m = elu(flat @ w_bidi^T + b) : 128x2048 @ 2048x1024 WMMA GEMMs
  gemm_wmma_kernel<<<1024 / 16, 256, 0, stream>>>(hFlat16, wBidi16, b_bidi,
                                                  h_m, 4 * Hc, 1024, 1);
  gemm_wmma_kernel<<<1024 / 16, 256, 0, stream>>>(cFlat16, wBidi16, b_bidi,
                                                  c_m, 4 * Hc, 1024, 1);

  // --- decoder init ---
  dec_init_kernel<<<pk, 256, 0, stream>>>(X, h_m, c_m, inp_y,
                                          hDec0, cDec0, hDec1, cDec1);

  // --- decoder loop ---
  for (int t = 0; t < HORc; ++t) {
    const int cur = t & 1, nxt = 1 - cur;
    xdec_kernel<<<(Bc * DINc + 255) / 256, 256, 0, stream>>>(inp_y, FF, t, xdec);
    // layer 0: tiny-K input via scalar FMA, recurrent via WMMA
    StepArgs a0{};
    a0.hIn  = hDec0 + (size_t)cur * BH;  a0.hInDirOff = 0;
    a0.wHH  = wD0hh16;                   a0.wHHDirOff = 0;
    a0.xA = nullptr; a0.wIH = nullptr;   a0.K2 = 0;
    a0.xs = xdec; a0.xsDirOff = 0;       a0.xsRowStride = DINc; a0.Ks = DINc;
    a0.wS = dec_w_ih_l0;                 a0.wSDirOff = 0;
    a0.bias = dec_b_l0;                  a0.biasDirOff = 0;
    a0.cBuf = cDec0;                     a0.cDirOff = 0;
    a0.hOut = hDec0 + (size_t)nxt * BH;  a0.hOutDirOff = 0;
    a0.ysOut = nullptr; a0.hOutF32 = nullptr;
    lstm_step_kernel<<<stepGrid1, 256, 0, stream>>>(a0);
    // layer 1: input chain = new h0 (K=512 WMMA), recurrent chain = h1
    StepArgs a1{};
    a1.hIn  = hDec1 + (size_t)cur * BH;  a1.hInDirOff = 0;
    a1.wHH  = wD1hh16;                   a1.wHHDirOff = 0;
    a1.xA   = hDec0 + (size_t)nxt * BH;  a1.xADirOff = 0;
    a1.xARowStride = Hc;                 a1.K2 = Hc;
    a1.wIH  = wD1ih16;                   a1.wIHDirOff = 0;
    a1.xs = nullptr; a1.wS = nullptr;    a1.Ks = 0;
    a1.bias = dec_b_l1;                  a1.biasDirOff = 0;
    a1.cBuf = cDec1;                     a1.cDirOff = 0;
    a1.hOut = hDec1 + (size_t)nxt * BH;  a1.hOutDirOff = 0;
    a1.ysOut = nullptr;
    a1.hOutF32 = hDec1F32;               a1.hOutF32DirOff = 0;
    lstm_step_kernel<<<stepGrid1, 256, 0, stream>>>(a1);
    // output head (N=5): scalar dot products; feeds next step's inp_y
    proj_kernel<<<(Bc * MQc + 255) / 256, 256, 0, stream>>>(
        hDec1F32, w_out, b_out, out, inp_y, t);
  }
}
